// MultiHeadAttention_46815143526502
// MI455X (gfx1250) — matrix operand order repaired and checked by reading, not compiled, not run on hardware
//
#include <hip/hip_runtime.h>
#include <hip/hip_bf16.h>

// ---------------------------------------------------------------------------
// MHA forward for gfx1250: bf16 WMMA GEMMs + fused flash-attention.
// B=2, S=2048, D_MODEL=1024, H=16, DK=64.
// ---------------------------------------------------------------------------

#define D_MODEL 1024
#define SEQ     2048
#define BATCH   2
#define NHEAD   16
#define DK      64
#define NTOK    (BATCH * SEQ)   // 4096

typedef __attribute__((ext_vector_type(16))) __bf16 v16bf;
typedef __attribute__((ext_vector_type(8)))  float  v8f;

// ---- bf16 helpers (raw u16 storage to avoid header type friction) ----------
__device__ __forceinline__ unsigned short f2bf(float f) {
    unsigned int x = __float_as_uint(f);
    unsigned int r = x + 0x7FFFu + ((x >> 16) & 1u);   // round-to-nearest-even
    return (unsigned short)(r >> 16);
}
__device__ __forceinline__ __bf16 bfb(unsigned short u) {
    union { unsigned short s; __bf16 b; } x; x.s = u; return x.b;
}
__device__ __forceinline__ v8f vzero8() {
    v8f z = {0.f, 0.f, 0.f, 0.f, 0.f, 0.f, 0.f, 0.f};
    return z;
}

// ---- 16-lane butterfly steps: DPP8 for xor 1/2/4, ds_swizzle for xor 8 -----
constexpr int dpp8_xor_sel(int m) {
    int s = 0;
    for (int i = 0; i < 8; ++i) s |= ((i ^ m) & 7) << (3 * i);
    return s;
}
template <int MASK>
__device__ __forceinline__ float bfly(float v) {
#if __has_builtin(__builtin_amdgcn_mov_dpp8)
    constexpr int sel = dpp8_xor_sel(MASK);
    return __int_as_float(__builtin_amdgcn_mov_dpp8(__float_as_int(v), sel));
#else
    return __shfl_xor(v, MASK, 32);
#endif
}
__device__ __forceinline__ float bfly8(float v) {
#if __has_builtin(__builtin_amdgcn_ds_swizzle)
    // group-of-32 swizzle: xor_mask=0x08, or_mask=0, and_mask=0x1f  (SWAPX8)
    return __int_as_float(
        __builtin_amdgcn_ds_swizzle(__float_as_int(v), (0x08 << 10) | 0x1f));
#else
    return __shfl_xor(v, 8, 32);
#endif
}

// A-fragment K index for 16x32 bf16 A tiles: contiguous k = 16*half + j (matches B)
__device__ __forceinline__ int a_kidx(int j, int hoff) {
    return 2 * hoff + j;
}

// ---------------------------------------------------------------------------
// GEMM: Y[M,N] = (X[M,K] @ W[N,K]^T + bias[N]) * out_scale   (M=4096,N=K=1024)
// Block tile 128x64, K-step 32, double-buffered LDS. 256 threads = 8 waves;
// wave w owns rows 16w..16w+15 and all 64 cols -> 4 WMMA C tiles.
// A tile is [m][k]; B tile is [n][k] so BOTH fragment gathers are
// lane-contiguous (2x ds_load_b128 each) -- no scalar LDS loads.
// OUT_TRANS writes Y transposed ([col][row]) for the V^T buffer.
// ---------------------------------------------------------------------------
template <bool X_BF16, bool OUT_BF16, bool OUT_TRANS>
__global__ __launch_bounds__(256)
void gemm_kernel(const void* __restrict__ Xv, const float* __restrict__ W,
                 const float* __restrict__ bias, void* __restrict__ Yv,
                 float out_scale)
{
    const int K = D_MODEL, N = D_MODEL;
    __shared__ unsigned short At[2][128][40];   // [buf][m][k]  (pad 32->40)
    __shared__ unsigned short Bt[2][64][40];    // [buf][n][k]  (pad 32->40)

    const int m0   = blockIdx.x * 128;
    const int n0   = blockIdx.y * 64;
    const int t    = threadIdx.x;
    const int lane = t & 31;
    const int w    = t >> 5;                    // 0..7 : 16-row group
    const int l16  = lane & 15;
    const int hoff = (lane < 16) ? 0 : 8;

    const float*          Xf = (const float*)Xv;
    const unsigned short* Xh = (const unsigned short*)Xv;

    // global->register staging (for LDS double buffering)
    const int am = t >> 1, ak = (t & 1) * 16;   // A: 16 contiguous k per thread
    const int bn = t >> 2, bk = (t & 3) * 8;    // B: 8 contiguous k per thread
    float          arf[16];
    unsigned short arh[16];
    float          brf[8];

    auto load_regs = [&](int k0) {
        if (X_BF16) {
            const unsigned short* src = Xh + (size_t)(m0 + am) * K + k0 + ak;
            #pragma unroll
            for (int i = 0; i < 16; ++i) arh[i] = src[i];
        } else {
            const float* src = Xf + (size_t)(m0 + am) * K + k0 + ak;
            #pragma unroll
            for (int i = 0; i < 16; ++i) arf[i] = src[i];
        }
        const float* bs = W + (size_t)(n0 + bn) * K + k0 + bk;
        #pragma unroll
        for (int i = 0; i < 8; ++i) brf[i] = bs[i];
    };
    auto store_regs = [&](int buf) {
        #pragma unroll
        for (int i = 0; i < 16; ++i)
            At[buf][am][ak + i] = X_BF16 ? arh[i] : f2bf(arf[i]);
        #pragma unroll
        for (int i = 0; i < 8; ++i)
            Bt[buf][bn][bk + i] = f2bf(brf[i]);     // contiguous store
    };

    v8f c[4];
    #pragma unroll
    for (int n2 = 0; n2 < 4; ++n2) c[n2] = vzero8();

    load_regs(0);
    store_regs(0);

    const int NSTEP = K / 32;
    for (int s = 0; s < NSTEP; ++s) {
        __syncthreads();
        if (s + 1 < NSTEP) load_regs((s + 1) * 32);   // overlap with compute
        const int cur = s & 1;

        // gather ALL fragments first (10 ds_load_b128 in flight), then WMMAs
        v16bf a, b[4];
        #pragma unroll
        for (int j = 0; j < 16; ++j)
            a[j] = bfb(At[cur][w * 16 + l16][a_kidx(j, hoff)]);
        #pragma unroll
        for (int n2 = 0; n2 < 4; ++n2) {
            const int kbase = (lane < 16) ? 0 : 16;   // B row is lane's column
            #pragma unroll
            for (int j = 0; j < 16; ++j)
                b[n2][j] = bfb(Bt[cur][n2 * 16 + l16][kbase + j]);
        }
        #pragma unroll
        for (int n2 = 0; n2 < 4; ++n2)
            c[n2] = __builtin_amdgcn_wmma_f32_16x16x32_bf16(false, a, false, b[n2],
                                                            (short)0, c[n2],
                                                            false, false);
        if (s + 1 < NSTEP) store_regs((s + 1) & 1);
    }

    // Epilogue: C layout row = r + 8*(lane>=16), col = l16
    #pragma unroll
    for (int n2 = 0; n2 < 4; ++n2) {
        #pragma unroll
        for (int r = 0; r < 8; ++r) {
            int row = m0 + w * 16 + r + hoff;
            int col = n0 + n2 * 16 + l16;
            float y = (c[n2][r] + bias[col]) * out_scale;
            if (OUT_TRANS)
                ((unsigned short*)Yv)[(size_t)col * NTOK + row] = f2bf(y);
            else if (OUT_BF16)
                ((unsigned short*)Yv)[(size_t)row * N + col] = f2bf(y);
            else
                ((float*)Yv)[(size_t)row * N + col] = y;
        }
    }
}

// ---------------------------------------------------------------------------
// Flash attention: block = 64 queries x one (batch,head); 4 waves x 16 queries.
// No LDS staging for K/V and no block barriers: K fragments load directly
// from row-major kb (lane = key row, d contiguous) and V fragments load
// directly from the TRANSPOSED vbufT (lane = d row, keys contiguous), both as
// 2x global_load_b128 per fragment; L2 (192MB >> 16MB of K/V) serves reuse.
// Per chunk per wave: 4 WMMAs (Q.K^T) + 5 WMMAs (P.[V|1]); ones-column WMMA
// accumulates the softmax denominator. Row-max butterfly: 3x DPP8 + swizzle.
// qb is pre-scaled by 1/sqrt(DK) in its GEMM epilogue.
// ---------------------------------------------------------------------------
__global__ __launch_bounds__(128)
void flash_attn_kernel(const unsigned short* __restrict__ qb,
                       const unsigned short* __restrict__ kb,
                       const unsigned short* __restrict__ vbT,
                       unsigned short* __restrict__ ctxb)
{
    __shared__ unsigned short Pst[4][16][40];    // per-wave P re-layout bounce

    const int bh    = blockIdx.y;        // b*NHEAD + h
    const int bidx  = bh >> 4;
    const int h     = bh & 15;
    const int qtile = blockIdx.x;
    const int t     = threadIdx.x;
    const int lane  = t & 31;
    const int w     = t >> 5;
    const int l16   = lane & 15;
    const int hoff  = (lane < 16) ? 0 : 8;
    const int kbase = (lane < 16) ? 0 : 16;      // B-fragment K offset per lane

    const size_t baserow = (size_t)bidx * SEQ;
    const int qrow0 = qtile * 64 + w * 16;
    const int colh  = h * DK;

    // Q fragments: 16 queries x 64 d -> two 16x32 bf16 A fragments
    v16bf qf[2];
    #pragma unroll
    for (int c = 0; c < 2; ++c) {
        const unsigned short* qp =
            qb + (baserow + qrow0 + l16) * D_MODEL + colh + c * 32;
        #pragma unroll
        for (int j = 0; j < 16; ++j) qf[c][j] = bfb(qp[a_kidx(j, hoff)]);
    }

    // ones-column B fragment (col 0 of a 16-wide tile = 1, rest 0)
    v16bf bones;
    #pragma unroll
    for (int j = 0; j < 16; ++j) bones[j] = bfb(l16 == 0 ? 0x3F80 : 0x0000);

    // per-lane fragment base pointers (advance by 32 keys per chunk)
    // K fragment (c,g): row = key (g*16 + l16), cols = c*32 + kbase + j
    const unsigned short* kp[2][2];
    #pragma unroll
    for (int c = 0; c < 2; ++c)
        #pragma unroll
        for (int g = 0; g < 2; ++g)
            kp[c][g] = kb + (baserow + g * 16 + l16) * D_MODEL
                          + colh + c * 32 + kbase;
    // V fragment (n2): row = dcol (colh + n2*16 + l16), cols = key
    const unsigned short* vp[4];
    #pragma unroll
    for (int n2 = 0; n2 < 4; ++n2)
        vp[n2] = vbT + (size_t)(colh + n2 * 16 + l16) * NTOK
                     + baserow + kbase;

    float m[8];
    v8f acc[5];                                  // [0..3]=ctx, [4]=row-sum l
    #pragma unroll
    for (int r = 0; r < 8; ++r) m[r] = -3.0e38f;
    #pragma unroll
    for (int n2 = 0; n2 < 5; ++n2) acc[n2] = vzero8();

    for (int kc = 0; kc < SEQ / 32; ++kc) {
        const size_t kadv = (size_t)kc * 32 * D_MODEL;   // K: 32 rows ahead
        const size_t vadv = (size_t)kc * 32;             // V^T: 32 cols ahead

        // ---- gather all 4 K^T fragments (8x global b128), then 4 WMMAs ----
        v16bf bkf[2][2];
        #pragma unroll
        for (int c = 0; c < 2; ++c)
            #pragma unroll
            for (int g = 0; g < 2; ++g)
                #pragma unroll
                for (int j = 0; j < 16; ++j)
                    bkf[c][g][j] = bfb(kp[c][g][kadv + j]);

        if (kc + 1 < SEQ / 32) {                 // global_prefetch_b8
            __builtin_prefetch(kp[0][0] + kadv + 32 * D_MODEL, 0, 1);
            __builtin_prefetch(vp[0] + vadv + 32, 0, 1);
        }

        v8f s0 = vzero8(), s1 = vzero8();
        s0 = __builtin_amdgcn_wmma_f32_16x16x32_bf16(false, qf[0], false, bkf[0][0],
                                                     (short)0, s0, false, false);
        s1 = __builtin_amdgcn_wmma_f32_16x16x32_bf16(false, qf[0], false, bkf[0][1],
                                                     (short)0, s1, false, false);
        s0 = __builtin_amdgcn_wmma_f32_16x16x32_bf16(false, qf[1], false, bkf[1][0],
                                                     (short)0, s0, false, false);
        s1 = __builtin_amdgcn_wmma_f32_16x16x32_bf16(false, qf[1], false, bkf[1][1],
                                                     (short)0, s1, false, false);

        // ---- issue V fragment loads now; latency hides under softmax VALU --
        v16bf bvf[4];
        #pragma unroll
        for (int n2 = 0; n2 < 4; ++n2)
            #pragma unroll
            for (int j = 0; j < 16; ++j)
                bvf[n2][j] = bfb(vp[n2][vadv + j]);

        // ---- online softmax: row max butterfly + rescale (sum via WMMA) ----
        #pragma unroll
        for (int r = 0; r < 8; ++r) {
            float x0 = s0[r], x1 = s1[r];
            float cm = fmaxf(x0, x1);
            cm = fmaxf(cm, bfly<1>(cm));
            cm = fmaxf(cm, bfly<2>(cm));
            cm = fmaxf(cm, bfly<4>(cm));
            cm = fmaxf(cm, bfly8(cm));
            float nm    = fmaxf(m[r], cm);
            float alpha = __expf(m[r] - nm);
            m[r] = nm;
            s0[r] = __expf(x0 - nm);
            s1[r] = __expf(x1 - nm);
            #pragma unroll
            for (int n2 = 0; n2 < 5; ++n2) acc[n2][r] *= alpha;
        }

        // ---- bounce P through per-wave LDS: C layout -> A layout ----
        #pragma unroll
        for (int r = 0; r < 8; ++r) {
            int row = r + hoff;
            Pst[w][row][l16]      = f2bf(s0[r]);
            Pst[w][row][16 + l16] = f2bf(s1[r]);
        }
        v16bf pa;
        #pragma unroll
        for (int j = 0; j < 16; ++j) pa[j] = bfb(Pst[w][l16][a_kidx(j, hoff)]);

        // ---- context += P @ V ; denominator += P @ ones ----
        #pragma unroll
        for (int n2 = 0; n2 < 4; ++n2)
            acc[n2] = __builtin_amdgcn_wmma_f32_16x16x32_bf16(false, pa, false, bvf[n2],
                                                              (short)0, acc[n2],
                                                              false, false);
        acc[4] = __builtin_amdgcn_wmma_f32_16x16x32_bf16(false, pa, false, bones,
                                                         (short)0, acc[4],
                                                         false, false);
    }

    // ---- finalize: l sits in col 0 of acc[4] (lanes 0 and 16) ----
    float inv[8];
    #pragma unroll
    for (int r = 0; r < 8; ++r) {
        float lsum = __shfl(acc[4][r], lane & 16, 32);   // broadcast within 16
        inv[r] = 1.0f / lsum;
    }
    #pragma unroll
    for (int n2 = 0; n2 < 4; ++n2) {
        #pragma unroll
        for (int r = 0; r < 8; ++r) {
            int row = qrow0 + r + hoff;
            int col = colh + n2 * 16 + l16;
            ctxb[(baserow + row) * D_MODEL + col] = f2bf(acc[n2][r] * inv[r]);
        }
    }
}

// ---------------------------------------------------------------------------
// Host side
// ---------------------------------------------------------------------------
extern "C" void kernel_launch(void* const* d_in, const int* in_sizes, int n_in,
                              void* d_out, int out_size, void* d_ws, size_t ws_size,
                              hipStream_t stream)
{
    (void)in_sizes; (void)n_in; (void)out_size; (void)ws_size;

    const float* query = (const float*)d_in[0];
    const float* key   = (const float*)d_in[1];
    const float* value = (const float*)d_in[2];
    const float* Wq    = (const float*)d_in[3];
    const float* bq    = (const float*)d_in[4];
    const float* Wk    = (const float*)d_in[5];
    const float* bk    = (const float*)d_in[6];
    const float* Wv    = (const float*)d_in[7];
    const float* bv    = (const float*)d_in[8];
    const float* Wo    = (const float*)d_in[9];
    const float* bo    = (const float*)d_in[10];

    // workspace: 4 raw-bf16 buffers (Q, K, V^T, ctx) = 32 MB
    const size_t NELT = (size_t)NTOK * D_MODEL;
    unsigned short* qbuf = (unsigned short*)d_ws;
    unsigned short* kbuf = qbuf + NELT;
    unsigned short* vbT  = kbuf + NELT;          // [D_MODEL][NTOK] transposed
    unsigned short* ctxb = vbT + NELT;

    dim3 gemm_grid(NTOK / 128, D_MODEL / 64);   // (32, 16)
    dim3 gemm_block(256);
    const float qscale = 0.125f;                // 1/sqrt(DK) folded into Q

    gemm_kernel<false, true, false><<<gemm_grid, gemm_block, 0, stream>>>(query, Wq, bq, qbuf, qscale);
    gemm_kernel<false, true, false><<<gemm_grid, gemm_block, 0, stream>>>(key,   Wk, bk, kbuf, 1.0f);
    gemm_kernel<false, true, true ><<<gemm_grid, gemm_block, 0, stream>>>(value, Wv, bv, vbT,  1.0f);

    dim3 fa_grid(SEQ / 64, BATCH * NHEAD);      // (32, 32)
    flash_attn_kernel<<<fa_grid, dim3(128), 0, stream>>>(qbuf, kbuf, vbT, ctxb);

    gemm_kernel<true, false, false><<<gemm_grid, gemm_block, 0, stream>>>(ctxb, Wo, bo, d_out, 1.0f);
}
